// HybridGNNModel_33071248180019
// MI455X (gfx1250) — matrix operand order repaired and checked by reading, not compiled, not run on hardware
//
#include <hip/hip_runtime.h>
#include <hip/hip_bf16.h>
#include <math.h>

#define NNODES 50000
#define NEDGES 800000
#define DK 64              // every GEMM in this model has K = 64
#define MT (NNODES / 16)   // 3125 exact 16-row tiles

typedef __attribute__((ext_vector_type(16))) __bf16 v16bf;
typedef __attribute__((ext_vector_type(16))) float  v16f;
typedef __attribute__((ext_vector_type(8)))  float  v8f;

__device__ __forceinline__ __bf16 f2bf(float f) {
  unsigned u = __float_as_uint(f);
  unsigned r = (u + 0x7FFFu + ((u >> 16) & 1u)) >> 16;   // round-to-nearest-even
  return __builtin_bit_cast(__bf16, (unsigned short)r);
}

// ---------------------------------------------------------------------------
// Weight prep: W[k][NC] (f32) -> WT[c][64] (bf16) so each WMMA B-fragment is a
// single contiguous 32B load per lane.
// ---------------------------------------------------------------------------
__global__ void prep_wt_kernel(const float* __restrict__ W, __bf16* __restrict__ WT, int NC) {
  int i = blockIdx.x * blockDim.x + threadIdx.x;
  if (i >= NC * DK) return;
  int c = i / DK, kk = i % DK;
  WT[(size_t)c * DK + kk] = f2bf(W[(size_t)kk * NC + c]);
}

// ---------------------------------------------------------------------------
// WMMA GEMM: C[M][NC] = A[M][64](f32, hw-cvt->bf16) x WT[NC][64](bf16) + bias
// 8 waves / block; each wave owns one 16-row tile and sweeps all NC/16 column
// tiles, reusing its two A fragments (K-steps 0 and 1) across every WMMA.
// A layout (16-bit, 16x32): lane<16 -> koff 0, lane>=16 -> koff 16;
//   halves 0..15 = K[kb+koff .. +15]
// B layout: lanes 0-15 hold K 0..15, lanes 16-31 hold K 16..31, N = lane&15.
// ---------------------------------------------------------------------------
__global__ void wmma_gemm_k64_kernel(const float* __restrict__ A,
                                     const __bf16* __restrict__ WT,
                                     const float* __restrict__ bias,
                                     float* __restrict__ C, int NC) {
  const int wave    = threadIdx.x >> 5;
  const int rowTile = blockIdx.x * 8 + wave;
  if (rowTile >= MT) return;                 // wave-uniform exit (EXEC stays all-1)
  const int lane  = threadIdx.x & 31;
  const int hi    = (lane >> 4) & 1;
  const int l15   = lane & 15;
  const int row   = rowTile * 16 + l15;
  const float* arow = A + (size_t)row * DK;
  const int koffA = hi ? 16 : 0;
  const int koffB = hi ? 16 : 0;

  // Build both A fragments once; reuse across all column tiles.
  v16bf afrag[2];
#pragma unroll
  for (int s = 0; s < 2; ++s) {
    const int kb = s * 32;
    const float4 a0 = *(const float4*)(arow + kb + koffA);
    const float4 a1 = *(const float4*)(arow + kb + koffA + 4);
    const float4 a2 = *(const float4*)(arow + kb + koffA + 8);
    const float4 a3 = *(const float4*)(arow + kb + koffA + 12);
    v16f af;
    af[0]  = a0.x; af[1]  = a0.y; af[2]  = a0.z; af[3]  = a0.w;
    af[4]  = a1.x; af[5]  = a1.y; af[6]  = a1.z; af[7]  = a1.w;
    af[8]  = a2.x; af[9]  = a2.y; af[10] = a2.z; af[11] = a2.w;
    af[12] = a3.x; af[13] = a3.y; af[14] = a3.z; af[15] = a3.w;
    afrag[s] = __builtin_convertvector(af, v16bf);   // hw v_cvt f32->bf16
  }

  const int nTiles = NC >> 4;
  for (int ct = 0; ct < nTiles; ++ct) {
    const int col = ct * 16 + l15;
    const __bf16* brow = WT + (size_t)col * DK;
    v8f acc = {};
#pragma unroll
    for (int s = 0; s < 2; ++s) {
      v16bf b = *(const v16bf*)(brow + s * 32 + koffB);   // 32B aligned
      acc = __builtin_amdgcn_wmma_f32_16x16x32_bf16(false, afrag[s], false, b,
                                                    (short)0, acc, false, false);
    }
    const float bval = bias ? bias[col] : 0.0f;
#pragma unroll
    for (int r = 0; r < 8; ++r) {
      int m = rowTile * 16 + r + (hi ? 8 : 0);   // C/D layout: VGPR r -> M
      C[(size_t)m * NC + col] = acc[r] + bval;
    }
  }
}

// ---------------------------------------------------------------------------
// Elementwise helpers
// ---------------------------------------------------------------------------
__global__ void fill_f32_kernel(float* __restrict__ p, int n, float v) {
  int i = blockIdx.x * blockDim.x + threadIdx.x;
  if (i < n) p[i] = v;
}

__global__ void relu_kernel(float* __restrict__ p, int n) {
  int i = blockIdx.x * blockDim.x + threadIdx.x;
  if (i < n) { float v = p[i]; p[i] = v > 0.0f ? v : 0.0f; }
}

__device__ __forceinline__ void atomicMaxF(float* addr, float val) {
  if (val >= 0.0f) atomicMax((int*)addr, __float_as_int(val));
  else             atomicMin((unsigned int*)addr, (unsigned int)__float_as_int(val));
}

// ---------------------------------------------------------------------------
// Phase 1: CEWGAT edge ops (wave32: lane + lane+32 covers 64 channels)
// ---------------------------------------------------------------------------
__global__ void gat_logit_kernel(const float* __restrict__ h,
                                 const float* __restrict__ att,
                                 const int* __restrict__ src,
                                 float* __restrict__ logit, int E) {
  int t = blockIdx.x * blockDim.x + threadIdx.x;
  int e = t >> 5, lane = t & 31;
  if (e >= E) return;
  const float* hr = h + (size_t)src[e] * 64;
  float v = hr[lane] * att[lane] + hr[lane + 32] * att[lane + 32];
#pragma unroll
  for (int off = 16; off > 0; off >>= 1) v += __shfl_xor(v, off, 32);
  if (lane == 0) logit[e] = (v > 0.0f) ? v : 0.2f * v;   // leaky_relu
}

__global__ void segmax_kernel(const float* __restrict__ logit,
                              const int* __restrict__ dst,
                              float* __restrict__ m, int E, int H) {
  int t = blockIdx.x * blockDim.x + threadIdx.x;
  if (t >= E * H) return;
  int e = t / H, hh = t - e * H;
  atomicMaxF(&m[(size_t)dst[e] * H + hh], logit[t]);
}

__global__ void segexp_kernel(float* __restrict__ logit,           // in: logit, out: exp
                              const int* __restrict__ dst,
                              const float* __restrict__ m,
                              float* __restrict__ z, int E, int H) {
  int t = blockIdx.x * blockDim.x + threadIdx.x;
  if (t >= E * H) return;
  int e = t / H, hh = t - e * H;
  size_t si = (size_t)dst[e] * H + hh;
  float a = expf(logit[t] - m[si]);
  logit[t] = a;
  atomicAdd(&z[si], a);
}

__global__ void gat_agg_kernel(const float* __restrict__ h,
                               const float* __restrict__ alpha,    // exp values
                               const float* __restrict__ z,
                               const float* __restrict__ cew,
                               const int* __restrict__ src,
                               const int* __restrict__ dst,
                               float* __restrict__ h1, int E) {
  int t = blockIdx.x * blockDim.x + threadIdx.x;
  int e = t >> 5, lane = t & 31;
  if (e >= E) return;
  int s = src[e], d = dst[e];
  float coef = alpha[e] / (z[d] + 1e-16f) * cew[e];
  const float* hr = h + (size_t)s * 64;
  float* o = h1 + (size_t)d * 64;
  atomicAdd(&o[lane],      hr[lane]      * coef);
  atomicAdd(&o[lane + 32], hr[lane + 32] * coef);
}

// ---------------------------------------------------------------------------
// Phase 2: TransformerConv edge ops (HEADS = 2, per-head dim 64)
// ---------------------------------------------------------------------------
__global__ void trans_logit_kernel(const float* __restrict__ q,
                                   const float* __restrict__ k,
                                   const int* __restrict__ src,
                                   const int* __restrict__ dst,
                                   float* __restrict__ logit2, int E) {
  int t = blockIdx.x * blockDim.x + threadIdx.x;
  int e = t >> 5, lane = t & 31;
  if (e >= E) return;
  const float* qd = q + (size_t)dst[e] * 128;
  const float* ks = k + (size_t)src[e] * 128;
#pragma unroll
  for (int hh = 0; hh < 2; ++hh) {
    int b = hh * 64;
    float v = qd[b + lane] * ks[b + lane] + qd[b + lane + 32] * ks[b + lane + 32];
#pragma unroll
    for (int off = 16; off > 0; off >>= 1) v += __shfl_xor(v, off, 32);
    if (lane == 0) logit2[(size_t)e * 2 + hh] = v * 0.125f;   // 1/sqrt(64)
  }
}

__global__ void trans_agg_kernel(const float* __restrict__ vmat,
                                 const float* __restrict__ alpha2,  // exp values
                                 const float* __restrict__ z2,
                                 const int* __restrict__ src,
                                 const int* __restrict__ dst,
                                 float* __restrict__ agg, int E) {
  int t = blockIdx.x * blockDim.x + threadIdx.x;
  int e = t >> 5, lane = t & 31;
  if (e >= E) return;
  int s = src[e], d = dst[e];
  const float* vs = vmat + (size_t)s * 128;
  float* o = agg + (size_t)d * 128;
#pragma unroll
  for (int hh = 0; hh < 2; ++hh) {
    float a = alpha2[(size_t)e * 2 + hh] / (z2[(size_t)d * 2 + hh] + 1e-16f);
    atomicAdd(&o[hh * 64 + lane],      vs[hh * 64 + lane]      * a);
    atomicAdd(&o[hh * 64 + lane + 32], vs[hh * 64 + lane + 32] * a);
  }
}

__global__ void combine_kernel(const float* __restrict__ agg,
                               const float* __restrict__ skip,
                               float* __restrict__ h2, int n) {
  int i = blockIdx.x * blockDim.x + threadIdx.x;
  if (i >= n) return;
  int node = i >> 6, c = i & 63;
  float a = 0.5f * (agg[(size_t)node * 128 + c] + agg[(size_t)node * 128 + 64 + c]);
  float v = a + skip[i];
  h2[i] = v > 0.0f ? v : 0.0f;
}

// ---------------------------------------------------------------------------
// Launch
// ---------------------------------------------------------------------------
extern "C" void kernel_launch(void* const* d_in, const int* in_sizes, int n_in,
                              void* d_out, int out_size, void* d_ws, size_t ws_size,
                              hipStream_t stream) {
  (void)in_sizes; (void)n_in; (void)out_size; (void)ws_size;
  const float* x     = (const float*)d_in[0];
  const float* cew   = (const float*)d_in[1];
  const float* W_gat = (const float*)d_in[2];
  const float* b_gat = (const float*)d_in[3];
  const float* att   = (const float*)d_in[4];
  const float* Wq    = (const float*)d_in[5];
  const float* bq    = (const float*)d_in[6];
  const float* Wk    = (const float*)d_in[7];
  const float* bk    = (const float*)d_in[8];
  const float* Wv    = (const float*)d_in[9];
  const float* bv    = (const float*)d_in[10];
  const float* Wskip = (const float*)d_in[11];
  const float* bskip = (const float*)d_in[12];
  const float* Wfc   = (const float*)d_in[13];
  const float* bfc   = (const float*)d_in[14];
  const int*   ei    = (const int*)d_in[15];
  const int*   src   = ei;            // edge_index row 0
  const int*   dst   = ei + NEDGES;   // edge_index row 1
  float* out = (float*)d_out;

  // workspace bump allocator (256B aligned)
  char* p = (char*)d_ws;
  auto alloc = [&p](size_t bytes) -> void* {
    void* r = (void*)p;
    p += (bytes + 255) & ~(size_t)255;
    return r;
  };
  __bf16* WTgat  = (__bf16*)alloc(64  * 64 * sizeof(__bf16));
  __bf16* WTq    = (__bf16*)alloc(128 * 64 * sizeof(__bf16));
  __bf16* WTk    = (__bf16*)alloc(128 * 64 * sizeof(__bf16));
  __bf16* WTv    = (__bf16*)alloc(128 * 64 * sizeof(__bf16));
  __bf16* WTskip = (__bf16*)alloc(64  * 64 * sizeof(__bf16));
  __bf16* WTfc   = (__bf16*)alloc(32  * 64 * sizeof(__bf16));
  float* h      = (float*)alloc((size_t)NNODES * 64 * 4);
  float* h1     = (float*)alloc((size_t)NNODES * 64 * 4);
  float* q      = (float*)alloc((size_t)NNODES * 128 * 4);
  float* k      = (float*)alloc((size_t)NNODES * 128 * 4);
  float* v      = (float*)alloc((size_t)NNODES * 128 * 4);
  float* skip   = (float*)alloc((size_t)NNODES * 64 * 4);
  float* logit1 = (float*)alloc((size_t)NEDGES * 4);
  float* logit2 = (float*)alloc((size_t)NEDGES * 2 * 4);
  float* m1 = (float*)alloc((size_t)NNODES * 4);
  float* z1 = (float*)alloc((size_t)NNODES * 4);
  float* m2 = (float*)alloc((size_t)NNODES * 2 * 4);
  float* z2 = (float*)alloc((size_t)NNODES * 2 * 4);
  float* agg = q;   // q,k dead after trans_logit; reuse
  float* h2  = h;   // h dead after gat_agg; reuse

  const int BLK = 256;
  const float NEG_INF = -__builtin_huge_valf();
  auto gz = [](int n, int b) { return (n + b - 1) / b; };
  const int ewarps = NEDGES * 32;              // warp-per-edge thread count
  const int gemmBlocks = (MT + 7) / 8;         // 8 waves (row tiles) per block

  // 0) weight prep
  hipLaunchKernelGGL(prep_wt_kernel, dim3(gz(64 * 64, BLK)),  dim3(BLK), 0, stream, W_gat, WTgat, 64);
  hipLaunchKernelGGL(prep_wt_kernel, dim3(gz(128 * 64, BLK)), dim3(BLK), 0, stream, Wq, WTq, 128);
  hipLaunchKernelGGL(prep_wt_kernel, dim3(gz(128 * 64, BLK)), dim3(BLK), 0, stream, Wk, WTk, 128);
  hipLaunchKernelGGL(prep_wt_kernel, dim3(gz(128 * 64, BLK)), dim3(BLK), 0, stream, Wv, WTv, 128);
  hipLaunchKernelGGL(prep_wt_kernel, dim3(gz(64 * 64, BLK)),  dim3(BLK), 0, stream, Wskip, WTskip, 64);
  hipLaunchKernelGGL(prep_wt_kernel, dim3(gz(32 * 64, BLK)),  dim3(BLK), 0, stream, Wfc, WTfc, 32);

  // 1) h = x @ W_gat + b_gat   (WMMA)
  hipLaunchKernelGGL(wmma_gemm_k64_kernel, dim3(gemmBlocks), dim3(BLK), 0, stream, x, WTgat, b_gat, h, 64);

  // 2) GAT softmax + scatter aggregation
  hipLaunchKernelGGL(fill_f32_kernel, dim3(gz(NNODES, BLK)), dim3(BLK), 0, stream, m1, NNODES, NEG_INF);
  hipLaunchKernelGGL(fill_f32_kernel, dim3(gz(NNODES, BLK)), dim3(BLK), 0, stream, z1, NNODES, 0.0f);
  hipLaunchKernelGGL(fill_f32_kernel, dim3(gz(NNODES * 64, BLK)), dim3(BLK), 0, stream, h1, NNODES * 64, 0.0f);
  hipLaunchKernelGGL(gat_logit_kernel, dim3(gz(ewarps, BLK)), dim3(BLK), 0, stream, h, att, src, logit1, NEDGES);
  hipLaunchKernelGGL(segmax_kernel, dim3(gz(NEDGES, BLK)), dim3(BLK), 0, stream, logit1, dst, m1, NEDGES, 1);
  hipLaunchKernelGGL(segexp_kernel, dim3(gz(NEDGES, BLK)), dim3(BLK), 0, stream, logit1, dst, m1, z1, NEDGES, 1);
  hipLaunchKernelGGL(gat_agg_kernel, dim3(gz(ewarps, BLK)), dim3(BLK), 0, stream,
                     h, logit1, z1, cew, src, dst, h1, NEDGES);
  hipLaunchKernelGGL(relu_kernel, dim3(gz(NNODES * 64, BLK)), dim3(BLK), 0, stream, h1, NNODES * 64);

  // 3) q/k/v/skip projections (WMMA)
  hipLaunchKernelGGL(wmma_gemm_k64_kernel, dim3(gemmBlocks), dim3(BLK), 0, stream, h1, WTq, bq, q, 128);
  hipLaunchKernelGGL(wmma_gemm_k64_kernel, dim3(gemmBlocks), dim3(BLK), 0, stream, h1, WTk, bk, k, 128);
  hipLaunchKernelGGL(wmma_gemm_k64_kernel, dim3(gemmBlocks), dim3(BLK), 0, stream, h1, WTv, bv, v, 128);
  hipLaunchKernelGGL(wmma_gemm_k64_kernel, dim3(gemmBlocks), dim3(BLK), 0, stream, h1, WTskip, bskip, skip, 64);

  // 4) transformer attention + scatter aggregation
  hipLaunchKernelGGL(fill_f32_kernel, dim3(gz(NNODES * 2, BLK)), dim3(BLK), 0, stream, m2, NNODES * 2, NEG_INF);
  hipLaunchKernelGGL(fill_f32_kernel, dim3(gz(NNODES * 2, BLK)), dim3(BLK), 0, stream, z2, NNODES * 2, 0.0f);
  hipLaunchKernelGGL(trans_logit_kernel, dim3(gz(ewarps, BLK)), dim3(BLK), 0, stream, q, k, src, dst, logit2, NEDGES);
  hipLaunchKernelGGL(fill_f32_kernel, dim3(gz(NNODES * 128, BLK)), dim3(BLK), 0, stream,
                     agg, NNODES * 128, 0.0f);   // after trans_logit: agg aliases q
  hipLaunchKernelGGL(segmax_kernel, dim3(gz(NEDGES * 2, BLK)), dim3(BLK), 0, stream, logit2, dst, m2, NEDGES, 2);
  hipLaunchKernelGGL(segexp_kernel, dim3(gz(NEDGES * 2, BLK)), dim3(BLK), 0, stream, logit2, dst, m2, z2, NEDGES, 2);
  hipLaunchKernelGGL(trans_agg_kernel, dim3(gz(ewarps, BLK)), dim3(BLK), 0, stream,
                     v, logit2, z2, src, dst, agg, NEDGES);

  // 5) h2 = relu(mean(heads) + skip)   (h2 aliases h)
  hipLaunchKernelGGL(combine_kernel, dim3(gz(NNODES * 64, BLK)), dim3(BLK), 0, stream,
                     agg, skip, h2, NNODES * 64);

  // 6) out = h2 @ Wfc + bfc   (WMMA, writes d_out directly)
  hipLaunchKernelGGL(wmma_gemm_k64_kernel, dim3(gemmBlocks), dim3(BLK), 0, stream, h2, WTfc, bfc, out, 32);
}
